// deepcod_21285857919606
// MI455X (gfx1250) — matrix operand order repaired and checked by reading, not compiled, not run on hardware
//
#include <hip/hip_runtime.h>
#include <hip/hip_bf16.h>
#include <math.h>

// ---------------------------------------------------------------------------
// Types for CDNA5 WMMA (wave32, 16x16x32 bf16 -> f32)
// ---------------------------------------------------------------------------
typedef __attribute__((ext_vector_type(16))) __bf16 v16bf;
typedef __attribute__((ext_vector_type(8)))  __bf16 v8bf;
typedef __attribute__((ext_vector_type(8)))  float  v8f;

union V16U { v16bf v; v8bf h[2]; __bf16 e[16]; };

__device__ __forceinline__ __bf16 f2bf(float f) {
  unsigned u = __builtin_bit_cast(unsigned, f);
  u += 0x7fffu + ((u >> 16) & 1u);                 // round-to-nearest-even
  unsigned short s = (unsigned short)(u >> 16);
  return __builtin_bit_cast(__bf16, s);
}

__device__ __forceinline__ unsigned hash_u32(unsigned x) {
  x ^= x >> 16; x *= 0x7feb352dU;
  x ^= x >> 15; x *= 0x846ca68bU;
  x ^= x >> 16; return x;
}

// ---------------------------------------------------------------------------
// Plain batched GEMM:  C[b][m][n] = sum_k A[b][m][k] * Bt[b][n][k]  (+bias, act)
// A: bf16 [Mp x Kp] rows padded; Bt: bf16 [Np16 x Kp]; C: f32 [M x N] (real dims)
// act: 0 none, 1 leaky(0.01), 2 tanh
// ---------------------------------------------------------------------------
__global__ void gemm_wmma(const __bf16* __restrict__ A, const __bf16* __restrict__ Bt,
                          const float* __restrict__ bias, float* __restrict__ C,
                          int M, int N, int Kp, long sA, long sB, long sC,
                          int tilesN, int tilesTotal, int act)
{
  int wave = threadIdx.x >> 5;
  int lane = threadIdx.x & 31;
  int tile = blockIdx.x * (blockDim.x >> 5) + wave;
  if (tile >= tilesTotal) return;                      // wave-uniform
  int tm = tile / tilesN, tn = tile % tilesN;
  int bz = blockIdx.z;
  const __bf16* Ab = A  + (long)bz * sA;
  const __bf16* Bb = Bt + (long)bz * sB;
  float*        Cb = C  + (long)bz * sC;

  int row = lane & 15, hi = lane >> 4;
  const __bf16* ap = Ab + (long)(tm * 16 + row) * Kp + hi * 8;
  const __bf16* bp = Bb + (long)(tn * 16 + row) * Kp + hi * 8;

  v8f acc = {0.f,0.f,0.f,0.f,0.f,0.f,0.f,0.f};
  for (int k0 = 0; k0 < Kp; k0 += 32) {
    V16U av, bv;
    av.h[0] = *(const v8bf*)(ap + k0);
    av.h[1] = *(const v8bf*)(ap + k0 + 16);
    bv.h[0] = *(const v8bf*)(bp + k0);
    bv.h[1] = *(const v8bf*)(bp + k0 + 16);
    acc = __builtin_amdgcn_wmma_f32_16x16x32_bf16(false, av.v, false, bv.v,
                                                  (short)0, acc, false, false);
  }
  int col = tn * 16 + row;
  float bsv = (bias && col < N) ? bias[col] : 0.f;
#pragma unroll
  for (int r = 0; r < 8; ++r) {
    int m = tm * 16 + r + 8 * hi;
    if (m < M && col < N) {
      float v = acc[r] + bsv;
      if (act == 1)      v = (v >= 0.f) ? v : 0.01f * v;
      else if (act == 2) v = tanhf(v);
      Cb[(long)m * N + col] = v;
    }
  }
}

// ---------------------------------------------------------------------------
// Implicit-GEMM convolution (channel-last activations, forward & transposed):
//   src: f32 [B,H,W,Cin];  Bt: bf16 [Np16 x Kp] (weights, K = Cin*KH*KW)
//   out: f32 [M x N], M = B*OH*OW (row = (b,oh,ow)), N = Cout
//   iy = oh*stride - pad + kh; valid iff iy%dil==0 and iy/dil in [0,H)
// ---------------------------------------------------------------------------
__device__ __forceinline__ float conv_fetch(const float* __restrict__ src,
    int b, int oh, int ow, int kk, bool mok,
    int K, int KHW, int KW, int Cin, int H, int W, int stride, int pad, int dil)
{
  if (!mok || kk >= K) return 0.f;
  int c = kk / KHW; int r = kk % KHW; int kh = r / KW; int kw = r % KW;
  int iy = oh * stride - pad + kh;
  int ix = ow * stride - pad + kw;
  if (iy < 0 || ix < 0) return 0.f;
  if (dil > 1) { if ((iy % dil) || (ix % dil)) return 0.f; iy /= dil; ix /= dil; }
  if (iy >= H || ix >= W) return 0.f;
  return src[(((long)b * H + iy) * W + ix) * Cin + c];
}

__global__ void conv_gemm_wmma(const float* __restrict__ src, const __bf16* __restrict__ Bt,
                               const float* __restrict__ bias, float* __restrict__ C,
                               int Cin, int H, int W, int OH, int OW, int KH, int KW,
                               int stride, int pad, int dil, int Kp, int M, int N,
                               int tilesN, int tilesTotal, int act)
{
  int wave = threadIdx.x >> 5;
  int lane = threadIdx.x & 31;
  int tile = blockIdx.x * (blockDim.x >> 5) + wave;
  if (tile >= tilesTotal) return;                      // wave-uniform
  int tm = tile / tilesN, tn = tile % tilesN;
  int row = lane & 15, hi = lane >> 4;

  int m = tm * 16 + row;
  bool mok = (m < M);
  int ow = 0, oh = 0, bb = 0;
  if (mok) { ow = m % OW; int t = m / OW; oh = t % OH; bb = t / OH; }
  int KHW = KH * KW;
  int K   = Cin * KHW;

  const __bf16* bp = Bt + (long)(tn * 16 + row) * Kp + hi * 8;

  v8f acc = {0.f,0.f,0.f,0.f,0.f,0.f,0.f,0.f};
  for (int k0 = 0; k0 < Kp; k0 += 32) {
    V16U av, bv;
#pragma unroll
    for (int j = 0; j < 8; ++j) {
      int kkA = k0 + hi * 8 + j;
      int kkB = k0 + 16 + hi * 8 + j;
      av.e[j]     = f2bf(conv_fetch(src, bb, oh, ow, kkA, mok, K, KHW, KW, Cin, H, W, stride, pad, dil));
      av.e[8 + j] = f2bf(conv_fetch(src, bb, oh, ow, kkB, mok, K, KHW, KW, Cin, H, W, stride, pad, dil));
    }
    bv.h[0] = *(const v8bf*)(bp + k0);
    bv.h[1] = *(const v8bf*)(bp + k0 + 16);
    acc = __builtin_amdgcn_wmma_f32_16x16x32_bf16(false, av.v, false, bv.v,
                                                  (short)0, acc, false, false);
  }
  int col = tn * 16 + row;
  float bsv = (bias && col < N) ? bias[col] : 0.f;
#pragma unroll
  for (int r = 0; r < 8; ++r) {
    int mm = tm * 16 + r + 8 * hi;
    if (mm < M && col < N) {
      float v = acc[r] + bsv;
      if (act == 1)      v = (v >= 0.f) ? v : 0.01f * v;
      else if (act == 2) v = tanhf(v);
      C[(long)mm * N + col] = v;
    }
  }
}

// ---------------------------------------------------------------------------
// Weight / layout prep kernels
// ---------------------------------------------------------------------------
// forward conv OIHW -> Bt[o][k] bf16, padded to [Op x Kp]
__global__ void k_prep_w(const float* __restrict__ w, __bf16* __restrict__ Bt,
                         int O, int K, int Kp, long total)
{
  long i = (long)blockIdx.x * blockDim.x + threadIdx.x;
  if (i >= total) return;
  int k = (int)(i % Kp); long o = i / Kp;
  float v = (o < O && k < K) ? w[o * (long)K + k] : 0.f;
  Bt[i] = f2bf(v);
}

// transposed conv [I,O,KH,KW] -> flipped forward kernel Bt[o][(i*KH+kh)*KW+kw]
__global__ void k_prep_wt(const float* __restrict__ w, __bf16* __restrict__ Bt,
                          int I, int O, int KH, int KW, int Kp, long total)
{
  long i = (long)blockIdx.x * blockDim.x + threadIdx.x;
  if (i >= total) return;
  int k = (int)(i % Kp); long o = i / Kp;
  int KHW = KH * KW; int K = I * KHW;
  float v = 0.f;
  if (o < O && k < K) {
    int c = k / KHW; int r = k % KHW; int kh = r / KW; int kw = r % KW;
    v = w[(((long)c * O + o) * KH + (KH - 1 - kh)) * KW + (KW - 1 - kw)];
  }
  Bt[i] = f2bf(v);
}

// f32 [Bc,R,S] -> bf16 [Bc,Rp,Sp], zero padded
__global__ void k_cast_pad(const float* __restrict__ src, __bf16* __restrict__ dst,
                           int Bc, int R, int S, int Rp, int Sp)
{
  long total = (long)Bc * Rp * Sp;
  long i = (long)blockIdx.x * blockDim.x + threadIdx.x;
  if (i >= total) return;
  int s = (int)(i % Sp); long t = i / Sp; int r = (int)(t % Rp); int b = (int)(t / Rp);
  float v = (r < R && s < S) ? src[((long)b * R + r) * S + s] : 0.f;
  dst[i] = f2bf(v);
}

// f32 [Bc,R,S] -> bf16 [Bc,S,Rp] transpose, zero padded in R
__global__ void k_transpose_pad(const float* __restrict__ src, __bf16* __restrict__ dst,
                                int Bc, int R, int S, int Rp)
{
  long total = (long)Bc * S * Rp;
  long i = (long)blockIdx.x * blockDim.x + threadIdx.x;
  if (i >= total) return;
  int r = (int)(i % Rp); long t = i / Rp; int s = (int)(t % S); int b = (int)(t / S);
  float v = (r < R) ? src[((long)b * R + r) * S + s] : 0.f;
  dst[i] = f2bf(v);
}

// ---------------------------------------------------------------------------
// Elementwise / reduction kernels
// ---------------------------------------------------------------------------
__global__ void k_quantize(const float* __restrict__ x, const float* __restrict__ centers,
                           float* __restrict__ q, long n)
{
  long i = (long)blockIdx.x * blockDim.x + threadIdx.x;
  if (i >= n) return;
  float v = x[i];
  float best = centers[0], bd = fabsf(v - best);
#pragma unroll
  for (int j = 1; j < 8; ++j) {
    float c = centers[j]; float d = fabsf(v - c);
    if (d < bd) { bd = d; best = c; }
  }
  q[i] = best;
}

__global__ void k_zero1(float* p) { p[0] = 0.f; }

__global__ void k_sumsq(const float* __restrict__ x, float* __restrict__ out, long n)
{
  __shared__ float red[256];
  long i = (long)blockIdx.x * blockDim.x + threadIdx.x;
  float a = 0.f;
  for (long j = i; j < n; j += (long)gridDim.x * blockDim.x) { float v = x[j]; a += v * v; }
  red[threadIdx.x] = a; __syncthreads();
  for (int s = 128; s > 0; s >>= 1) {
    if (threadIdx.x < s) red[threadIdx.x] += red[threadIdx.x + s];
    __syncthreads();
  }
  if (threadIdx.x == 0) atomicAdd(out, red[0]);
}

__global__ void k_noise(const float* __restrict__ xq, float* __restrict__ dst,
                        const float* __restrict__ sumsq, const int* __restrict__ snr, long n)
{
  long i = (long)blockIdx.x * blockDim.x + threadIdx.x;
  if (i >= n) return;
  float sigp = sumsq[0] / (float)n;
  float std  = sqrtf(sigp / powf(10.f, (float)snr[0] * 0.1f));
  unsigned h1 = hash_u32((unsigned)i * 2u + 1u);
  unsigned h2 = hash_u32((unsigned)i * 2u + 2u);
  float u1 = ((h1 >> 8) + 1u) * (1.f / 16777217.f);
  float u2 = (h2 >> 8) * (1.f / 16777216.f);
  float g  = sqrtf(-2.f * logf(u1)) * cosf(6.2831853f * u2);
  dst[i] = xq[i] + std * g;
}

__global__ void k_add(float* __restrict__ dst, const float* __restrict__ a,
                      const float* __restrict__ b, long n)
{
  long i = (long)blockIdx.x * blockDim.x + threadIdx.x;
  if (i < n) dst[i] = a[i] + b[i];
}

__global__ void k_gamma_res(float* __restrict__ dst, const float* __restrict__ a,
                            const float* __restrict__ b, const float* __restrict__ gamma, long n)
{
  long i = (long)blockIdx.x * blockDim.x + threadIdx.x;
  if (i < n) dst[i] = gamma[0] * a[i] + b[i];
}

__global__ void k_copy(float* __restrict__ dst, const float* __restrict__ src, long n)
{
  long i = (long)blockIdx.x * blockDim.x + threadIdx.x;
  if (i < n) dst[i] = src[i];
}

// channel-last [B,HW,C] -> NCHW [B,C,HW]
__global__ void k_cl2nchw(const float* __restrict__ src, float* __restrict__ dst,
                          int B, int C, int HW)
{
  long n = (long)B * C * HW;
  long i = (long)blockIdx.x * blockDim.x + threadIdx.x;
  if (i >= n) return;
  int hw = (int)(i % HW); long t = i / HW; int c = (int)(t % C); int b = (int)(t / C);
  dst[i] = src[((long)b * HW + hw) * C + c];
}

// maxpool channel-last VALID
__global__ void k_maxpool_cl(const float* __restrict__ src, float* __restrict__ dst,
                             int B, int H, int W, int C, int k, int s, int OH, int OW)
{
  long n = (long)B * OH * OW * C;
  long i = (long)blockIdx.x * blockDim.x + threadIdx.x;
  if (i >= n) return;
  int c = (int)(i % C); long t = i / C;
  int ow = (int)(t % OW); t /= OW;
  int oh = (int)(t % OH); int b = (int)(t / OH);
  float m = -3.4e38f;
  for (int kh = 0; kh < k; ++kh)
    for (int kw = 0; kw < k; ++kw) {
      float v = src[(((long)b * H + oh * s + kh) * W + ow * s + kw) * C + c];
      m = fmaxf(m, v);
    }
  dst[i] = m;
}

// softmax over last dim; one block per row
__global__ void k_softmax(float* __restrict__ x, int rows, int cols)
{
  int r = blockIdx.x;
  if (r >= rows) return;
  float* p = x + (long)r * cols;
  __shared__ float red[256];
  int tid = threadIdx.x;
  float m = -3.4e38f;
  for (int i = tid; i < cols; i += blockDim.x) m = fmaxf(m, p[i]);
  red[tid] = m; __syncthreads();
  for (int s = 128; s > 0; s >>= 1) { if (tid < s) red[tid] = fmaxf(red[tid], red[tid + s]); __syncthreads(); }
  m = red[0]; __syncthreads();
  float sum = 0.f;
  for (int i = tid; i < cols; i += blockDim.x) { float e = expf(p[i] - m); p[i] = e; sum += e; }
  red[tid] = sum; __syncthreads();
  for (int s = 128; s > 0; s >>= 1) { if (tid < s) red[tid] += red[tid + s]; __syncthreads(); }
  float inv = 1.f / red[0];
  __syncthreads();
  for (int i = tid; i < cols; i += blockDim.x) p[i] *= inv;
}

// batchnorm over channel-last rows: stats per channel
__global__ void k_bn_stats(const float* __restrict__ x, float* __restrict__ mv, long rows, int C)
{
  int c = blockIdx.x;
  __shared__ float s1[256], s2[256];
  int tid = threadIdx.x;
  float a = 0.f, b = 0.f;
  for (long r = tid; r < rows; r += blockDim.x) {
    float v = x[r * C + c]; a += v; b += v * v;
  }
  s1[tid] = a; s2[tid] = b; __syncthreads();
  for (int s = 128; s > 0; s >>= 1) {
    if (tid < s) { s1[tid] += s1[tid + s]; s2[tid] += s2[tid + s]; }
    __syncthreads();
  }
  if (tid == 0) {
    float m = s1[0] / (float)rows;
    mv[c] = m; mv[C + c] = s2[0] / (float)rows - m * m;
  }
}

__global__ void k_bn_apply(float* __restrict__ x, const float* __restrict__ mv,
                           const float* __restrict__ g, const float* __restrict__ b,
                           int C, long n, int leaky)
{
  long i = (long)blockIdx.x * blockDim.x + threadIdx.x;
  if (i >= n) return;
  int c = (int)(i % C);
  float v = (x[i] - mv[c]) * rsqrtf(mv[C + c] + 1e-5f) * g[c] + b[c];
  if (leaky) v = (v >= 0.f) ? v : 0.01f * v;
  x[i] = v;
}

// ---------------------------------------------------------------------------
// Host side
// ---------------------------------------------------------------------------
static inline int cdiv_i(int a, int b) { return (a + b - 1) / b; }
static inline long cdiv_l(long a, long b) { return (a + b - 1) / b; }

struct ConvG { int Cin,H,W,OH,OW,KH,KW,stride,pad,dil,Kp,M,N; };

static void launch_conv(hipStream_t st, const float* src, const __bf16* Bt,
                        const float* bias, float* C, ConvG g, int act)
{
  int tilesM = cdiv_i(g.M, 16), tilesN = cdiv_i(g.N, 16);
  int tiles = tilesM * tilesN;
  conv_gemm_wmma<<<dim3(cdiv_i(tiles, 4)), dim3(128), 0, st>>>(
      src, Bt, bias, C, g.Cin, g.H, g.W, g.OH, g.OW, g.KH, g.KW,
      g.stride, g.pad, g.dil, g.Kp, g.M, g.N, tilesN, tiles, act);
}

static void launch_gemm(hipStream_t st, const __bf16* A, const __bf16* Bt,
                        const float* bias, float* C, int M, int N, int Kp,
                        long sA, long sB, long sC, int batches, int act)
{
  int tilesM = cdiv_i(M, 16), tilesN = cdiv_i(N, 16);
  int tiles = tilesM * tilesN;
  gemm_wmma<<<dim3(cdiv_i(tiles, 4), 1, batches), dim3(128), 0, st>>>(
      A, Bt, bias, C, M, N, Kp, sA, sB, sC, tilesN, tiles, act);
}

#define ELGRID(n) dim3((unsigned)cdiv_l((n), 256)), dim3(256), 0, stream

extern "C" void kernel_launch(void* const* d_in, const int* in_sizes, int n_in,
                              void* d_out, int out_size, void* d_ws, size_t ws_size,
                              hipStream_t stream)
{
  // --- input pointers (setup_inputs() recursive dict order) ---
  const float* in      = (const float*)d_in[0];   // [16,3136,64] channel-last
  const float* centers = (const float*)d_in[1];   // [8]
  const float* conv_w  = (const float*)d_in[2];  const float* conv_b = (const float*)d_in[3];
  const float* a1_qw = (const float*)d_in[4];  const float* a1_qb = (const float*)d_in[5];
  const float* a1_kw = (const float*)d_in[6];  const float* a1_kb = (const float*)d_in[7];
  const float* a1_vw = (const float*)d_in[8];  const float* a1_vb = (const float*)d_in[9];
  const float* a1_aw = (const float*)d_in[10]; const float* a1_ab = (const float*)d_in[11];
  const float* a1_g  = (const float*)d_in[12];
  const float* r1_w1 = (const float*)d_in[13]; const float* r1_b1 = (const float*)d_in[14];
  const float* r1_g1 = (const float*)d_in[15]; const float* r1_e1 = (const float*)d_in[16];
  const float* r1_w2 = (const float*)d_in[17]; const float* r1_b2 = (const float*)d_in[18];
  const float* r1_g2 = (const float*)d_in[19]; const float* r1_e2 = (const float*)d_in[20];
  const float* r1_w3 = (const float*)d_in[21]; const float* r1_b3 = (const float*)d_in[22];
  const float* a2_qw = (const float*)d_in[23]; const float* a2_qb = (const float*)d_in[24];
  const float* a2_kw = (const float*)d_in[25]; const float* a2_kb = (const float*)d_in[26];
  const float* a2_vw = (const float*)d_in[27]; const float* a2_vb = (const float*)d_in[28];
  const float* a2_aw = (const float*)d_in[29]; const float* a2_ab = (const float*)d_in[30];
  const float* a2_g  = (const float*)d_in[31];
  const float* r2_w1 = (const float*)d_in[32]; const float* r2_b1 = (const float*)d_in[33];
  const float* r2_g1 = (const float*)d_in[34]; const float* r2_e1 = (const float*)d_in[35];
  const float* r2_w2 = (const float*)d_in[36]; const float* r2_b2 = (const float*)d_in[37];
  const float* r2_g2 = (const float*)d_in[38]; const float* r2_e2 = (const float*)d_in[39];
  const float* r2_w3 = (const float*)d_in[40]; const float* r2_b3 = (const float*)d_in[41];
  const float* bn_g  = (const float*)d_in[42]; const float* bn_b  = (const float*)d_in[43];
  const float* out_w = (const float*)d_in[44]; const float* out_b = (const float*)d_in[45];
  const int*   snr   = (const int*)  d_in[(n_in > 46) ? 46 : n_in - 1];
  float* out = (float*)d_out;

  const int B = 16;
  // --- bump allocator in d_ws ---
  char* base = (char*)d_ws; size_t off = 0;
  auto alloc = [&](size_t bytes) -> void* {
    off = (off + 255) & ~(size_t)255;
    void* p = base + off; off += bytes; return p;
  };

  __bf16* wbt = (__bf16*)alloc(512 * 1024);               // reused weight Bt buffer
  float*  mv  = (float*) alloc(2 * 128 * sizeof(float));  // bn mean/var
  float*  scl = (float*) alloc(256);                      // sumsq scalar

  // ===== conv1: [16,56,56,64] -> [16,28,28,32], k2 s2 =====
  float* y1 = (float*)alloc((size_t)12544 * 32 * 4);
  {
    long t = 32L * 256; k_prep_w<<<ELGRID(t)>>>(conv_w, wbt, 32, 256, 256, t);
    ConvG g = {64,56,56,28,28,2,2,2,0,1,256,12544,32};
    launch_conv(stream, in, wbt, conv_b, y1, g, 0);
  }

  // ===== quantize + AWGN =====
  const long NQ = 12544L * 32;  // 401408
  float* xq  = (float*)alloc(NQ * 4);
  float* cwn = (float*)alloc(NQ * 4);
  k_quantize<<<ELGRID(NQ)>>>(y1, centers, xq, NQ);
  k_zero1<<<dim3(1), dim3(1), 0, stream>>>(scl);
  k_sumsq<<<dim3(512), dim3(256), 0, stream>>>(xq, scl, NQ);
  k_noise<<<ELGRID(NQ)>>>(xq, cwn, scl, snr, NQ);

  // ===== self-attention 1 (C=32, cq=3, pool 4/4 on 28x28, Np=49) =====
  float* a1out;
  {
    float* q1 = (float*)alloc((size_t)16 * 784 * 3 * 4);
    float* k1 = (float*)alloc((size_t)16 * 784 * 3 * 4);
    float* v1 = (float*)alloc((size_t)16 * 784 * 32 * 4);
    ConvG g1x1 = {32,28,28,28,28,1,1,1,0,1,32,12544,3};
    long tw = 16L * 32; k_prep_w<<<ELGRID(tw)>>>(a1_qw, wbt, 3, 32, 32, tw);
    launch_conv(stream, cwn, wbt, a1_qb, q1, g1x1, 0);
    k_prep_w<<<ELGRID(tw)>>>(a1_kw, wbt, 3, 32, 32, tw);
    launch_conv(stream, cwn, wbt, a1_kb, k1, g1x1, 0);
    long tv = 32L * 32; k_prep_w<<<ELGRID(tv)>>>(a1_vw, wbt, 32, 32, 32, tv);
    ConvG gv = g1x1; gv.N = 32;
    launch_conv(stream, cwn, wbt, a1_vb, v1, gv, 0);

    float* pq = (float*)alloc((size_t)16 * 49 * 3 * 4);
    float* pv = (float*)alloc((size_t)16 * 49 * 32 * 4);
    k_maxpool_cl<<<ELGRID(16L*49*3)>>>(q1, pq, B, 28, 28, 3, 4, 4, 7, 7);
    k_maxpool_cl<<<ELGRID(16L*49*32)>>>(v1, pv, B, 28, 28, 32, 4, 4, 7, 7);

    __bf16* aq = (__bf16*)alloc((size_t)16 * 64 * 32 * 2);
    __bf16* bk = (__bf16*)alloc((size_t)16 * 784 * 32 * 2);
    k_cast_pad<<<ELGRID(16L*64*32)>>>(pq, aq, 16, 49, 3, 64, 32);
    k_cast_pad<<<ELGRID(16L*784*32)>>>(k1, bk, 16, 784, 3, 784, 32);

    float* eng = (float*)alloc((size_t)16 * 49 * 784 * 4);
    launch_gemm(stream, aq, bk, nullptr, eng, 49, 784, 32, 64*32, 784*32, 49*784, 16, 0);
    k_softmax<<<dim3(16 * 49), dim3(256), 0, stream>>>(eng, 16 * 49, 784);

    __bf16* at  = (__bf16*)alloc((size_t)16 * 784 * 64 * 2);
    __bf16* pvt = (__bf16*)alloc((size_t)16 * 32 * 64 * 2);
    k_transpose_pad<<<ELGRID(16L*784*64)>>>(eng, at, 16, 49, 784, 64);
    k_transpose_pad<<<ELGRID(16L*32*64)>>>(pv, pvt, 16, 49, 32, 64);

    float* o1 = (float*)alloc((size_t)16 * 784 * 32 * 4);
    launch_gemm(stream, at, pvt, nullptr, o1, 784, 32, 64, 784*64, 32*64, 784*32, 16, 0);

    a1out = (float*)alloc((size_t)16 * 784 * 32 * 4);
    k_prep_w<<<ELGRID(tv)>>>(a1_aw, wbt, 32, 32, 32, tv);
    launch_conv(stream, o1, wbt, a1_ab, a1out, gv, 0);
    k_gamma_res<<<ELGRID(NQ)>>>(a1out, a1out, cwn, a1_g, NQ);
  }

  // ===== res_t 1: 32ch 28x28 -> 64ch 56x56 =====
  const long N1 = 16L * 3136 * 64;  // 3,211,264
  float* h2;
  {
    float* h1  = (float*)alloc(N1 * 4);
    h2         = (float*)alloc(N1 * 4);
    float* sc  = (float*)alloc(N1 * 4);
    long t1 = 64L * 288;
    k_prep_wt<<<ELGRID(t1)>>>(r1_w1, wbt, 32, 64, 3, 3, 288, t1);
    ConvG gu = {32,28,28,56,56,3,3,1,1,2,288,50176,64};
    launch_conv(stream, a1out, wbt, r1_b1, h1, gu, 0);
    k_bn_stats<<<dim3(64), dim3(256), 0, stream>>>(h1, mv, 50176, 64);
    k_bn_apply<<<ELGRID(N1)>>>(h1, mv, r1_g1, r1_e1, 64, N1, 1);

    long t2 = 64L * 576;
    k_prep_wt<<<ELGRID(t2)>>>(r1_w2, wbt, 64, 64, 3, 3, 576, t2);
    ConvG gs = {64,56,56,56,56,3,3,1,1,1,576,50176,64};
    launch_conv(stream, h1, wbt, r1_b2, h2, gs, 0);
    k_bn_stats<<<dim3(64), dim3(256), 0, stream>>>(h2, mv, 50176, 64);
    k_bn_apply<<<ELGRID(N1)>>>(h2, mv, r1_g2, r1_e2, 64, N1, 1);

    k_prep_wt<<<ELGRID(t1)>>>(r1_w3, wbt, 32, 64, 3, 3, 288, t1);
    launch_conv(stream, a1out, wbt, r1_b3, sc, gu, 0);
    k_add<<<ELGRID(N1)>>>(h2, h2, sc, N1);   // rt1 output in h2: [16,56,56,64]
  }

  // ===== self-attention 2 (C=64, cq=8, pool 8/8 on 56x56, Np=49) =====
  float* a2out;
  {
    float* q2 = (float*)alloc((size_t)16 * 3136 * 8 * 4);
    float* k2 = (float*)alloc((size_t)16 * 3136 * 8 * 4);
    float* v2 = (float*)alloc(N1 * 4);
    ConvG g1x1 = {64,56,56,56,56,1,1,1,0,1,64,50176,8};
    long tw = 16L * 64; k_prep_w<<<ELGRID(tw)>>>(a2_qw, wbt, 8, 64, 64, tw);
    launch_conv(stream, h2, wbt, a2_qb, q2, g1x1, 0);
    k_prep_w<<<ELGRID(tw)>>>(a2_kw, wbt, 8, 64, 64, tw);
    launch_conv(stream, h2, wbt, a2_kb, k2, g1x1, 0);
    long tv = 64L * 64; k_prep_w<<<ELGRID(tv)>>>(a2_vw, wbt, 64, 64, 64, tv);
    ConvG gv = g1x1; gv.N = 64;
    launch_conv(stream, h2, wbt, a2_vb, v2, gv, 0);

    float* pq = (float*)alloc((size_t)16 * 49 * 8 * 4);
    float* pv = (float*)alloc((size_t)16 * 49 * 64 * 4);
    k_maxpool_cl<<<ELGRID(16L*49*8)>>>(q2, pq, B, 56, 56, 8, 8, 8, 7, 7);
    k_maxpool_cl<<<ELGRID(16L*49*64)>>>(v2, pv, B, 56, 56, 64, 8, 8, 7, 7);

    __bf16* aq = (__bf16*)alloc((size_t)16 * 64 * 32 * 2);
    __bf16* bk = (__bf16*)alloc((size_t)16 * 3136 * 32 * 2);
    k_cast_pad<<<ELGRID(16L*64*32)>>>(pq, aq, 16, 49, 8, 64, 32);
    k_cast_pad<<<ELGRID(16L*3136*32)>>>(k2, bk, 16, 3136, 8, 3136, 32);

    float* eng = (float*)alloc((size_t)16 * 49 * 3136 * 4);
    launch_gemm(stream, aq, bk, nullptr, eng, 49, 3136, 32, 64*32, 3136*32, 49*3136, 16, 0);
    k_softmax<<<dim3(16 * 49), dim3(256), 0, stream>>>(eng, 16 * 49, 3136);

    __bf16* at  = (__bf16*)alloc((size_t)16 * 3136 * 64 * 2);
    __bf16* pvt = (__bf16*)alloc((size_t)16 * 64 * 64 * 2);
    k_transpose_pad<<<ELGRID(16L*3136*64)>>>(eng, at, 16, 49, 3136, 64);
    k_transpose_pad<<<ELGRID(16L*64*64)>>>(pv, pvt, 16, 49, 64, 64);

    float* o2 = (float*)alloc(N1 * 4);
    launch_gemm(stream, at, pvt, nullptr, o2, 3136, 64, 64, 3136*64, 64*64, 3136*64, 16, 0);

    a2out = (float*)alloc(N1 * 4);
    k_prep_w<<<ELGRID(tv)>>>(a2_aw, wbt, 64, 64, 64, tv);
    launch_conv(stream, o2, wbt, a2_ab, a2out, gv, 0);
    k_gamma_res<<<ELGRID(N1)>>>(a2out, a2out, h2, a2_g, N1);
  }

  // ===== res_t 2: 64ch 56x56 -> 128ch 112x112, then outer BN+leaky =====
  const long N2 = 16L * 12544 * 128;  // 25,690,112
  float* h4;
  {
    float* h3 = (float*)alloc(N2 * 4);
    h4        = (float*)alloc(N2 * 4);
    float* sc = (float*)alloc(N2 * 4);
    long t1 = 128L * 576;
    k_prep_wt<<<ELGRID(t1)>>>(r2_w1, wbt, 64, 128, 3, 3, 576, t1);
    ConvG gu = {64,56,56,112,112,3,3,1,1,2,576,200704,128};
    launch_conv(stream, a2out, wbt, r2_b1, h3, gu, 0);
    k_bn_stats<<<dim3(128), dim3(256), 0, stream>>>(h3, mv, 200704, 128);
    k_bn_apply<<<ELGRID(N2)>>>(h3, mv, r2_g1, r2_e1, 128, N2, 1);

    long t2 = 128L * 1152;
    k_prep_wt<<<ELGRID(t2)>>>(r2_w2, wbt, 128, 128, 3, 3, 1152, t2);
    ConvG gs = {128,112,112,112,112,3,3,1,1,1,1152,200704,128};
    launch_conv(stream, h3, wbt, r2_b2, h4, gs, 0);
    k_bn_stats<<<dim3(128), dim3(256), 0, stream>>>(h4, mv, 200704, 128);
    k_bn_apply<<<ELGRID(N2)>>>(h4, mv, r2_g2, r2_e2, 128, N2, 1);

    k_prep_wt<<<ELGRID(t1)>>>(r2_w3, wbt, 64, 128, 3, 3, 576, t1);
    launch_conv(stream, a2out, wbt, r2_b3, sc, gu, 0);
    k_add<<<ELGRID(N2)>>>(h4, h4, sc, N2);

    k_bn_stats<<<dim3(128), dim3(256), 0, stream>>>(h4, mv, 200704, 128);
    k_bn_apply<<<ELGRID(N2)>>>(h4, mv, bn_g, bn_b, 128, N2, 1);
  }

  // ===== final conv: 128ch 112x112 -> 64ch 56x56 s2 p1, tanh, direct to d_out =====
  // d_out layout: [codeSent 401408][cwn NCHW 401408][codeReceived 3211264]
  //               [cwn NCHW 401408][x_q NCHW 401408]
  {
    long tw = 64L * 1152;
    k_prep_w<<<ELGRID(tw)>>>(out_w, wbt, 64, 1152, 1152, tw);
    ConvG g = {128,112,112,56,56,3,3,2,1,1,1152,50176,64};
    launch_conv(stream, h4, wbt, out_b, out + 802816, g, 2 /*tanh*/);
  }

  // ===== outputs =====
  k_copy<<<ELGRID(NQ)>>>(out + 0, xq, NQ);                                 // codeSent
  k_cl2nchw<<<ELGRID(NQ)>>>(cwn, out + 401408, B, 32, 784);                // codeWithNoise
  k_cl2nchw<<<ELGRID(NQ)>>>(cwn, out + 4014080, B, 32, 784);               // codeWithNoise (again)
  k_cl2nchw<<<ELGRID(NQ)>>>(xq,  out + 4415488, B, 32, 784);               // x_q

  (void)in_sizes; (void)out_size; (void)ws_size;
}
